// MoELayer_51178830299715
// MI455X (gfx1250) — compile-verified, matrix operand order checked by reading, not run on hardware
//
#include <hip/hip_runtime.h>
#include <hip/hip_bf16.h>

// ---------------------------------------------------------------- constants
constexpr int T    = 8192;   // B*S tokens
constexpr int D    = 1024;
constexpr int FF   = 4096;
constexpr int E    = 8;
constexpr int KTOP = 2;

constexpr int BM  = 128;     // token tile
constexpr int BN  = 128;     // output-feature tile
constexpr int BK  = 32;      // k tile (bf16 WMMA K=32)
constexpr int BKP = BK + 8;  // padded LDS stride in halves (80B: 16B aligned chunks, skewed banks)

typedef __attribute__((ext_vector_type(16))) __bf16 v16bf;
typedef __attribute__((ext_vector_type(8)))  __bf16 v8bf;
typedef __attribute__((ext_vector_type(8)))  float  v8f;
typedef __attribute__((ext_vector_type(4)))  float  v4f;
typedef __attribute__((ext_vector_type(4)))  int    v4i;

union FragU { v16bf v; v8bf h[2]; };

// ---------------------------------------------------------------- scheduler pinning
#if defined(__has_builtin)
#if __has_builtin(__builtin_amdgcn_sched_group_barrier)
#define SCHED_GROUP(mask, cnt) __builtin_amdgcn_sched_group_barrier(mask, cnt, 0)
#endif
#endif
#ifndef SCHED_GROUP
#define SCHED_GROUP(mask, cnt)
#endif

// ---------------------------------------------------------------- async global->LDS (CDNA5)
#if defined(__has_builtin)
#if __has_builtin(__builtin_amdgcn_global_load_async_to_lds_b128) && \
    __has_builtin(__builtin_amdgcn_s_wait_asynccnt)
#define MOE_ASYNC 1
#endif
#endif

#ifdef MOE_ASYNC
typedef __attribute__((address_space(1))) v4i gv4i;
typedef __attribute__((address_space(3))) v4i lv4i;
__device__ __forceinline__ void async_cp16(const void* g, void* l) {
  __builtin_amdgcn_global_load_async_to_lds_b128((gv4i*)g, (lv4i*)l, 0, 0);
}
__device__ __forceinline__ void async_wait0() { __builtin_amdgcn_s_wait_asynccnt(0); }
#else
__device__ __forceinline__ void async_wait0() {}
#endif

// A/B fragment for v_wmma_f32_16x16x32_bf16 (ISA 7.12.2, 16-bit 16x32):
// lane L: row = L&15; halves 0..7 hold K = 8*(L>>4)+i, halves 8..15 hold K = 16+8*(L>>4)+i.
__device__ __forceinline__ v16bf load_frag(const __bf16* base, int lane, int stride) {
  const __bf16* p = base + (size_t)(lane & 15) * stride + ((lane >> 4) << 3);
  FragU f;
  f.h[0] = *(const v8bf*)(p);
  f.h[1] = *(const v8bf*)(p + 16);
  return f.v;
}

// ---------------------------------------------------------------- init
__global__ __launch_bounds__(256) void moe_init(float* __restrict__ out, int* __restrict__ counts) {
  size_t i = (size_t)blockIdx.x * 256 + threadIdx.x;   // grid covers T*D/4
  v4f z = {0.f, 0.f, 0.f, 0.f};
  ((v4f*)out)[i] = z;
  if (blockIdx.x == 0 && threadIdx.x < 16) counts[threadIdx.x] = 0;
}

// ---------------------------------------------------------------- gating
// one wave (32 lanes, wave32) per token; w_gate (8x1024 f32 = 32KB) staged in LDS
__global__ __launch_bounds__(256) void moe_gate(
    const float* __restrict__ x, const float* __restrict__ wg, const float* __restrict__ bg,
    float* __restrict__ outIdx, float* __restrict__ outVal,
    int* __restrict__ counts, int* __restrict__ tokIdx, float* __restrict__ gateW)
{
  __shared__ float wgs[E * D];
  for (int i = threadIdx.x; i < E * D; i += 256) wgs[i] = wg[i];
  __syncthreads();

  const int wave = threadIdx.x >> 5, lane = threadIdx.x & 31;
  const int t = blockIdx.x * 8 + wave;

  float p[E];
#pragma unroll
  for (int e = 0; e < E; ++e) p[e] = 0.f;
  const float* xr = x + (size_t)t * D;
  for (int d = lane; d < D; d += 32) {
    float xv = xr[d];
#pragma unroll
    for (int e = 0; e < E; ++e) p[e] += xv * wgs[e * D + d];
  }
#pragma unroll
  for (int off = 16; off > 0; off >>= 1)
#pragma unroll
    for (int e = 0; e < E; ++e) p[e] += __shfl_xor(p[e], off, 32);

  if (lane == 0) {
    float mx = -1e30f;
#pragma unroll
    for (int e = 0; e < E; ++e) { p[e] += bg[e]; mx = fmaxf(mx, p[e]); }
    float s = 0.f;
#pragma unroll
    for (int e = 0; e < E; ++e) { p[e] = __expf(p[e] - mx); s += p[e]; }
    float inv = 1.f / s;
#pragma unroll
    for (int e = 0; e < E; ++e) p[e] *= inv;
    int i0 = 0; float v0 = p[0];
#pragma unroll
    for (int e = 1; e < E; ++e) if (p[e] > v0) { v0 = p[e]; i0 = e; }
    int i1 = 0; float v1 = -1.f;
#pragma unroll
    for (int e = 0; e < E; ++e) if (e != i0 && p[e] > v1) { v1 = p[e]; i1 = e; }
    outIdx[t * 2 + 0] = (float)i0; outIdx[t * 2 + 1] = (float)i1;
    outVal[t * 2 + 0] = v0;        outVal[t * 2 + 1] = v1;
    int s0 = atomicAdd(&counts[i0], 1); tokIdx[i0 * T + s0] = t; gateW[i0 * T + s0] = v0;
    int s1 = atomicAdd(&counts[i1], 1); tokIdx[i1 * T + s1] = t; gateW[i1 * T + s1] = v1;
  }
}

// ---------------------------------------------------------------- expert linear1 (+bias+leakyrelu)
// h[slot, f] = leaky(sum_d X[tok(slot), d] * W1[e][f, d] + b1[e][f]) as bf16
// double-buffered LDS; sched_group_barrier pins VMEM->DSread->WMMA->DSwrite pipeline
__global__ __launch_bounds__(256) void moe_h(
    const float* __restrict__ x, const float* __restrict__ w1, const float* __restrict__ b1,
    const int* __restrict__ counts, const int* __restrict__ tokIdx,
    __bf16* __restrict__ hBuf, int e)
{
  const int cnt = counts[e];
  const int m0 = blockIdx.x * BM;
  if (m0 >= cnt) return;
  const int n0 = blockIdx.y * BN;
  const float* w1e = w1 + (size_t)e * FF * D;
  const float* b1e = b1 + (size_t)e * FF;
  const int* toks = tokIdx + e * T;

  __shared__ __attribute__((aligned(16))) __bf16 As[2][BM][BKP];
  __shared__ __attribute__((aligned(16))) __bf16 Bs[2][BN][BKP];
  __shared__ int tokS[BM];

  const int tid = threadIdx.x;
  if (tid < BM) tokS[tid] = (m0 + tid < cnt) ? toks[m0 + tid] : 0;
  __syncthreads();

  const int wave = tid >> 5, lane = tid & 31;
  v8f acc[8];
#pragma unroll
  for (int n = 0; n < 8; ++n)
#pragma unroll
    for (int r = 0; r < 8; ++r) acc[n][r] = 0.f;

  const int rowL = tid >> 1;          // 0..127
  const int colL = (tid & 1) * 16;    // 0 / 16
  const float* aSrcRow = x + (size_t)tokS[rowL] * D + colL;
  const float* bSrcRow = w1e + (size_t)(n0 + rowL) * D + colL;

  float ta[16], tb[16];
  auto fetch = [&](int k0) {
    const float* as = aSrcRow + k0;
    const float* bs = bSrcRow + k0;
    __builtin_prefetch(bs + BK, 0, 1);
#pragma unroll
    for (int i = 0; i < 4; ++i) {
      *(v4f*)(ta + 4 * i) = *(const v4f*)(as + 4 * i);
      *(v4f*)(tb + 4 * i) = *(const v4f*)(bs + 4 * i);
    }
  };
  auto stage = [&](int buf) {
    __bf16* ad = &As[buf][rowL][colL];
    __bf16* bd = &Bs[buf][rowL][colL];
#pragma unroll
    for (int i = 0; i < 16; ++i) { ad[i] = (__bf16)ta[i]; bd[i] = (__bf16)tb[i]; }
  };

  fetch(0);
  stage(0);
  __syncthreads();

  constexpr int NK = D / BK;
  for (int k = 0; k < NK; ++k) {
    const int buf = k & 1;
    if (k + 1 < NK) fetch((k + 1) * BK);           // global loads for next tile in flight
    v16bf a = load_frag(&As[buf][wave * 16][0], lane, BKP);
    v16bf bfrag[8];
#pragma unroll
    for (int n = 0; n < 8; ++n) bfrag[n] = load_frag(&Bs[buf][n * 16][0], lane, BKP);
#pragma unroll
    for (int n = 0; n < 8; ++n)
      acc[n] = __builtin_amdgcn_wmma_f32_16x16x32_bf16(false, a, false, bfrag[n],
                                                       (short)0, acc[n], false, false);
    if (k + 1 < NK) stage(buf ^ 1);                // convert + ds_store behind the WMMA burst
    // pin the pipeline: VMEM reads early, DS reads clustered, WMMAs back-to-back, DS writes last
    SCHED_GROUP(0x020, 8);    // VMEM read: next-tile global_load_b128 x8
    SCHED_GROUP(0x100, 18);   // DS read:  A frag x2 + B frags x16
    SCHED_GROUP(0x008, 8);    // WMMA burst
    SCHED_GROUP(0x200, 4);    // DS write: stage next tile
    __syncthreads();
  }

  // C layout: VGPR r -> M = r + 8*(lane>=16); N = lane&15
  const int mBase = m0 + wave * 16 + (lane >> 4) * 8;
  const int nCol = lane & 15;
#pragma unroll
  for (int n = 0; n < 8; ++n) {
    int f = n0 + n * 16 + nCol;
    float bias = b1e[f];
#pragma unroll
    for (int r = 0; r < 8; ++r) {
      int m = mBase + r;
      if (m < cnt) {
        float hv = acc[n][r] + bias;
        hv = hv > 0.f ? hv : 0.1f * hv;
        hBuf[(size_t)m * FF + f] = (__bf16)hv;
      }
    }
  }
}

// ---------------------------------------------------------------- expert linear2 (+bias, gate, scatter-add)
// out[tok, dcol] += gate * (sum_f h[slot,f] * W2[e][dcol,f] + b2[e][dcol])
// A operand (bf16 hBuf) goes global->LDS via async DMA when available.
__global__ __launch_bounds__(256) void moe_y(
    const __bf16* __restrict__ hBuf, const float* __restrict__ w2, const float* __restrict__ b2,
    const int* __restrict__ counts, const int* __restrict__ tokIdx, const float* __restrict__ gateW,
    float* __restrict__ out, int e)
{
  const int cnt = counts[e];
  const int m0 = blockIdx.x * BM;
  if (m0 >= cnt) return;
  const int n0 = blockIdx.y * BN;
  const float* w2e = w2 + (size_t)e * D * FF;
  const float* b2e = b2 + (size_t)e * D;
  const int* toks = tokIdx + e * T;
  const float* gws = gateW + e * T;

  __shared__ __attribute__((aligned(16))) __bf16 As[2][BM][BKP];
  __shared__ __attribute__((aligned(16))) __bf16 Bs[2][BN][BKP];

  const int tid = threadIdx.x;
  const int wave = tid >> 5, lane = tid & 31;
  v8f acc[8];
#pragma unroll
  for (int n = 0; n < 8; ++n)
#pragma unroll
    for (int r = 0; r < 8; ++r) acc[n][r] = 0.f;

  const int rowL = tid >> 1;
  const int colL = (tid & 1) * 16;
  const __bf16* aSrcRow = hBuf + (size_t)(m0 + rowL) * FF + colL;  // slot rows, in-bounds (<= T)
  const float* bSrcRow = w2e + (size_t)(n0 + rowL) * FF + colL;

  float tb[16];
#ifndef MOE_ASYNC
  v8bf taLo, taHi;
#endif
  auto fetchA = [&](int k0, int buf) {
    const __bf16* as = aSrcRow + k0;
#ifdef MOE_ASYNC
    async_cp16(as,     &As[buf][rowL][colL]);      // global_load_async_to_lds_b128
    async_cp16(as + 8, &As[buf][rowL][colL + 8]);
#else
    taLo = *(const v8bf*)(as);
    taHi = *(const v8bf*)(as + 8);
#endif
  };
  auto fetchB = [&](int k0) {
    const float* bs = bSrcRow + k0;
    __builtin_prefetch(bs + BK, 0, 1);
#pragma unroll
    for (int i = 0; i < 4; ++i) *(v4f*)(tb + 4 * i) = *(const v4f*)(bs + 4 * i);
  };
  auto stage = [&](int buf) {
#ifndef MOE_ASYNC
    *(v8bf*)&As[buf][rowL][colL] = taLo;
    *(v8bf*)&As[buf][rowL][colL + 8] = taHi;
#endif
    __bf16* bd = &Bs[buf][rowL][colL];
#pragma unroll
    for (int i = 0; i < 16; ++i) bd[i] = (__bf16)tb[i];
  };

  fetchA(0, 0);
  fetchB(0);
  stage(0);
  async_wait0();
  __syncthreads();

  constexpr int NK = FF / BK;
  for (int k = 0; k < NK; ++k) {
    const int buf = k & 1;
    if (k + 1 < NK) { fetchA((k + 1) * BK, buf ^ 1); fetchB((k + 1) * BK); }
    v16bf a = load_frag(&As[buf][wave * 16][0], lane, BKP);
    v16bf bfrag[8];
#pragma unroll
    for (int n = 0; n < 8; ++n) bfrag[n] = load_frag(&Bs[buf][n * 16][0], lane, BKP);
#pragma unroll
    for (int n = 0; n < 8; ++n)
      acc[n] = __builtin_amdgcn_wmma_f32_16x16x32_bf16(false, a, false, bfrag[n],
                                                       (short)0, acc[n], false, false);
    if (k + 1 < NK) stage(buf ^ 1);
    SCHED_GROUP(0x020, 4);    // VMEM read: next-tile W2 global_load_b128 x4
    SCHED_GROUP(0x100, 18);   // DS read: A frag x2 + B frags x16
    SCHED_GROUP(0x008, 8);    // WMMA burst
    SCHED_GROUP(0x200, 2);    // DS write: stage next B tile
    async_wait0();                                  // s_wait_asynccnt 0 before publishing buf^1
    __syncthreads();
  }

  const int mLocal = wave * 16 + (lane >> 4) * 8;
  const int nCol = lane & 15;
#pragma unroll
  for (int r = 0; r < 8; ++r) {
    int m = m0 + mLocal + r;
    if (m < cnt) {
      int tok = toks[m];
      float g = gws[m];
      float* orow = out + (size_t)tok * D;
#pragma unroll
      for (int n = 0; n < 8; ++n) {
        int dcol = n0 + n * 16 + nCol;
        float v = g * (acc[n][r] + b2e[dcol]);
        __hip_atomic_fetch_add(&orow[dcol], v, __ATOMIC_RELAXED, __HIP_MEMORY_SCOPE_AGENT);
      }
    }
  }
}

// ---------------------------------------------------------------- launch
extern "C" void kernel_launch(void* const* d_in, const int* in_sizes, int n_in,
                              void* d_out, int out_size, void* d_ws, size_t ws_size,
                              hipStream_t stream) {
  const float* x      = (const float*)d_in[0];
  const float* w_gate = (const float*)d_in[1];
  const float* b_gate = (const float*)d_in[2];
  const float* w1     = (const float*)d_in[3];
  const float* b1     = (const float*)d_in[4];
  const float* w2     = (const float*)d_in[5];
  const float* b2     = (const float*)d_in[6];

  float* out    = (float*)d_out;
  float* outIdx = out + (size_t)T * D;           // topk_idx (stored as float)
  float* outVal = outIdx + (size_t)T * KTOP;     // topk_vals

  // workspace layout
  int*    counts = (int*)d_ws;                   // E (16 slots for alignment)
  int*    tokIdx = counts + 16;                  // E*T compacted token lists
  float*  gateW  = (float*)(tokIdx + E * T);     // E*T gate weights per slot
  __bf16* hBuf   = (__bf16*)(gateW + E * T);     // T*FF bf16, reused per expert (~64MB)

  moe_init<<<(T * D) / (256 * 4), 256, 0, stream>>>(out, counts);
  moe_gate<<<T / 8, 256, 0, stream>>>(x, w_gate, b_gate, outIdx, outVal, counts, tokIdx, gateW);
  for (int e = 0; e < E; ++e) {
    moe_h<<<dim3(T / BM, FF / BN), 256, 0, stream>>>(x, w1, b1, counts, tokIdx, hBuf, e);
    moe_y<<<dim3(T / BM, D / BN), 256, 0, stream>>>(hBuf, w2, b2, counts, tokIdx, gateW, out, e);
  }
}
